// MLPConstructor_7172595384463
// MI455X (gfx1250) — matrix operand order repaired and checked by reading, not compiled, not run on hardware
//
#include <hip/hip_runtime.h>
#include <hip/hip_bf16.h>

typedef __attribute__((ext_vector_type(16))) _Float16 v16h;
typedef __attribute__((ext_vector_type(8)))  float    v8f;
typedef __attribute__((ext_vector_type(4)))  float    f32x4;

#define LATENT 32

// ---------------------------------------------------------------------------
// Phase 1: per-node scalar projections via V_WMMA_F32_16X16X32_F16.
// x: [R, 32] f32 node features.  phi: [R, 8] f32:
//   cols (0,1) = (rel, mask) weights for "x1 of same-type block"   (W[:D])
//   cols (2,3) = (rel, mask) weights for "x2 of same-type block"   (W[D:])
//   cols (4,5) = (rel, mask) weights for "x1 of cross-type block"  (W[:D])
//   cols (6,7) = (rel, mask) weights for "x2 of cross-type block"  (W[D:])
// spatial=1: same=ss(rel0), x1-cross=st(rel2), x2-cross=ts(rel3)
// spatial=0: same=tt(rel1), x1-cross=ts(rel3), x2-cross=st(rel2)
// ---------------------------------------------------------------------------
__global__ __launch_bounds__(256) void phi_wmma_kernel(
    const float* __restrict__ x,
    const float* __restrict__ Wrel,   // [4, 64]
    const float* __restrict__ Wmask,  // [4, 64]
    float* __restrict__ phi,          // [R, 8]
    int R, int spatial)
{
  const int lane    = threadIdx.x & 31;
  const int wave    = threadIdx.x >> 5;
  const int rowBase = (blockIdx.x * 8 + wave) * 16;
  if (rowBase + 16 > R) return;          // wave-uniform -> EXEC stays all-1s

  const int m  = lane & 15;
  const int hi = lane >> 4;              // 0: lanes 0-15, 1: lanes 16-31

  // ---- A tile (16x32, f16). Lane holds row m; K chunks per ISA layout. ----
  const float* xp    = x + (size_t)(rowBase + m) * LATENT;
  const int    kbase = hi ? 8 : 0;       // lanes>=16 start at K=8 (and K=24)
  v16h a;
  #pragma unroll
  for (int c = 0; c < 2; ++c) {
    const int   k0 = kbase + c * 16;     // {0,16} or {8,24}: 32B aligned
    const float4 f0 = *(const float4*)(xp + k0);
    const float4 f1 = *(const float4*)(xp + k0 + 4);
    a[c*8+0] = (_Float16)f0.x; a[c*8+1] = (_Float16)f0.y;
    a[c*8+2] = (_Float16)f0.z; a[c*8+3] = (_Float16)f0.w;
    a[c*8+4] = (_Float16)f1.x; a[c*8+5] = (_Float16)f1.y;
    a[c*8+6] = (_Float16)f1.z; a[c*8+7] = (_Float16)f1.w;
  }

  // ---- B tile (32x16, f16): column n = lane&15 (cols 8..15 are zero). ----
  const int n  = lane & 15;
  const int kb = hi ? 8 : 0;             // lanes 0-15: K=0..7,16..23; 16-31: K=8..15,24..31
  const int p  = (n >> 1) & 3;           // column pair -> (relation, half)
  int rel, half;
  if (spatial) { const int rm[4] = {0,0,2,3}; const int hm[4] = {0,1,0,1};
                 rel = rm[p]; half = hm[p]; }
  else         { const int rm[4] = {1,1,3,2}; const int hm[4] = {0,1,0,1};
                 rel = rm[p]; half = hm[p]; }
  const float* wp = ((n & 1) ? Wmask : Wrel) + rel * 64 + half * 32;
  v16h bmat;
  #pragma unroll
  for (int u = 0; u < 16; ++u)
    bmat[u] = (n < 8) ? (_Float16)wp[kb + (u & 7) + ((u >> 3) << 4)] : (_Float16)0.0f;

  // ---- D = A x B (+0) : 16x16 f32 accumulator ----
  v8f c = {};
  c = __builtin_amdgcn_wmma_f32_16x16x32_f16(
      /*neg_a=*/false, a, /*neg_b=*/false, bmat,
      /*c_mod=*/(short)0, c, /*reuse_a=*/false, /*reuse_b=*/false);

  // ---- D layout: VGPR v holds M = v + 8*hi, N = lane&15 ----
  if (n < 8) {
    #pragma unroll
    for (int v = 0; v < 8; ++v)
      phi[(size_t)(rowBase + hi * 8 + v) * 8 + n] = c[v];
  }
}

// ---------------------------------------------------------------------------
// Phase 2: out[b,i,j] = tanh((ra+ca+b_rel) * relu(rm+cm+b_mask)).
// One block per output row (b,i); 320 threads x float4 = 1280 columns.
// 209.7 MB of output -> non-temporal b128 stores; phi reads stay in L2.
// ---------------------------------------------------------------------------
__global__ __launch_bounds__(320) void adj_kernel(
    const float* __restrict__ phiS,   // [B*N, 8]
    const float* __restrict__ phiT,   // [B*T, 8]
    const float* __restrict__ brel,   // [4]
    const float* __restrict__ bmask,  // [4]
    float* __restrict__ out)          // [B, 1280, 1280]
{
  const int N = 1024, T = 256, M = N + T;
  const int row = blockIdx.x;              // 0 .. B*M-1
  const int b   = row / M;
  const int i   = row - b * M;
  const bool iS = (i < N);
  const float* pr = iS ? (phiS + (size_t)(b * N + i) * 8)
                       : (phiT + (size_t)(b * T + (i - N)) * 8);

  const int  j0 = threadIdx.x * 4;         // float4 never crosses j=1024
  const bool jS = (j0 < N);
  const int  r    = iS ? (jS ? 0 : 2) : (jS ? 3 : 1);   // ss / st / ts / tt
  const bool same = (iS == jS);
  const float ra = pr[same ? 0 : 4] + brel[r];
  const float rm = pr[same ? 1 : 5] + bmask[r];
  const int colOff = same ? 2 : 6;
  const float* pc = jS ? (phiS + (size_t)(b * N + j0) * 8)
                       : (phiT + (size_t)(b * T + (j0 - N)) * 8);

  f32x4 o;
  #pragma unroll
  for (int u = 0; u < 4; ++u) {
    const float2 cv = *(const float2*)(pc + u * 8 + colOff);  // 8B aligned
    const float a  = ra + cv.x;
    const float mv = fmaxf(rm + cv.y, 0.0f);
    o[u] = tanhf(a * mv);
  }
  __builtin_nontemporal_store(o, (f32x4*)(out + (size_t)row * M + j0));
}

// ---------------------------------------------------------------------------
extern "C" void kernel_launch(void* const* d_in, const int* in_sizes, int n_in,
                              void* d_out, int out_size, void* d_ws, size_t ws_size,
                              hipStream_t stream) {
  const float* s     = (const float*)d_in[0];  // [32,1024,32]
  const float* t     = (const float*)d_in[1];  // [32, 256,32]
  const float* Wrel  = (const float*)d_in[2];  // [4,64]
  const float* brel  = (const float*)d_in[3];  // [4]
  const float* Wmask = (const float*)d_in[4];  // [4,64]
  const float* bmask = (const float*)d_in[5];  // [4]
  float* out = (float*)d_out;

  const int Rs = 32 * 1024;   // spatial rows
  const int Rt = 32 * 256;    // temporal rows
  float* phiS = (float*)d_ws;            // 1.00 MB
  float* phiT = phiS + (size_t)Rs * 8;   // 0.25 MB  (ws total 1.25 MB)

  // 8 waves/block, 16 rows/wave -> 128 rows/block
  phi_wmma_kernel<<<Rs / 128, 256, 0, stream>>>(s, Wrel, Wmask, phiS, Rs, 1);
  phi_wmma_kernel<<<Rt / 128, 256, 0, stream>>>(t, Wrel, Wmask, phiT, Rt, 0);
  adj_kernel<<<32 * 1280, 320, 0, stream>>>(phiS, phiT, brel, bmask, out);
}
